// StressCondGNNLayer_85796266705417
// MI455X (gfx1250) — matrix operand order repaired and checked by reading, not compiled, not run on hardware
//
#include <hip/hip_runtime.h>
#include <hip/hip_bf16.h>
#include <math.h>

#define NODES 50000
#define EDGES 400000
#define HDIM  128
#define SDIM  64
#define HEADS 4
#define CCH   128          // per-head channels
#define XCOLS 512          // HEADS*CCH
#define NEG_SLOPE 0.2f
#define LN_EPS 1e-5f

typedef __attribute__((ext_vector_type(16))) _Float16 v16h;
typedef __attribute__((ext_vector_type(8)))  float    v8f;

__device__ __forceinline__ float warp_sum(float v) {
  #pragma unroll
  for (int m = 16; m >= 1; m >>= 1) v += __shfl_xor(v, m, 32);
  return v;
}
// order-preserving int encoding of float for integer atomicMax
__device__ __forceinline__ int   fenc(float f){ int i = __float_as_int(f); return (i >= 0) ? i : (i ^ 0x7fffffff); }
__device__ __forceinline__ float fdec(int e){ int i = (e >= 0) ? e : (e ^ 0x7fffffff); return __int_as_float(i); }
__device__ __forceinline__ float lrelu(float x){ return x > 0.f ? x : NEG_SLOPE * x; }

// ---------------- FiLM projection: g = gamma@Wp+bp, b = beta@Wp+bp ----------
__global__ void film_kernel(const float* __restrict__ gamma, const float* __restrict__ beta,
                            const float* __restrict__ Wp, const float* __restrict__ bp,
                            float* __restrict__ g, float* __restrict__ bf) {
  int t = threadIdx.x;
  int c = t & 127;
  int w = t >> 7;            // 0 -> gamma, 1 -> beta
  const float* v = w ? beta : gamma;
  float s = bp[c];
  #pragma unroll 8
  for (int k = 0; k < SDIM; ++k) s += v[k] * Wp[k * HDIM + c];
  (w ? bf : g)[c] = s;
}

// ---- pre-transpose + f16-convert weights so B fragments are one b256 load --
// WgT[col*128 + k] = f16(Wg[k*512 + col])   (512x128)
// W1T[col*256 + k] = f16(W1[k*128 + col])   (128x256)
__global__ void prep_weights_kernel(const float* __restrict__ Wg, const float* __restrict__ W1,
                                    _Float16* __restrict__ WgT, _Float16* __restrict__ W1T) {
  int tid = blockIdx.x * blockDim.x + threadIdx.x;       // 65536 threads
  {
    int col = tid >> 7, k = tid & 127;                   // Wg: 128x512 -> 512x128
    WgT[tid] = (_Float16)Wg[(size_t)k * XCOLS + col];
  }
  if (tid < 256 * HDIM) {
    int col = tid >> 8, k = tid & 255;                   // W1: 256x128 -> 128x256
    W1T[tid] = (_Float16)W1[(size_t)k * HDIM + col];
  }
}

// ---------------- GEMM1: xh = f16(h @ Wg)  [N,512] ---------------------------
__global__ void __launch_bounds__(256) gemm_x_kernel(const float* __restrict__ h,
                                                     const _Float16* __restrict__ WgT,
                                                     _Float16* __restrict__ xh) {
  __shared__ _Float16 hA[16 * HDIM];            // 16-node f16 tile of h
  const int row0 = blockIdx.x * 16;
  const int tid  = threadIdx.x;
  for (int i = tid; i < 16 * HDIM; i += 256)
    hA[i] = (_Float16)h[(size_t)(row0 + (i >> 7)) * HDIM + (i & 127)];
  __syncthreads();

  const int wave = tid >> 5, lane = tid & 31, half = lane >> 4, l16 = lane & 15;
  v8f acc[4] = { {}, {}, {}, {} };

  for (int k0 = 0; k0 < HDIM; k0 += 32) {
    if (k0 + 32 < HDIM)  // stream next K-slab of transposed weights
      __builtin_prefetch(&WgT[(size_t)(wave * 64 + l16) * HDIM + k0 + 32], 0, 1);
    v16h a;
    #pragma unroll
    for (int e2 = 0; e2 < 16; ++e2) {
      int k = k0 + half * 16 + e2;   // A 16x32 f16: this code packs 16 contiguous K per lane half (B matches)
      a[e2] = hA[l16 * HDIM + k];
    }
    #pragma unroll
    for (int t = 0; t < 4; ++t) {
      int col = wave * 64 + t * 16 + l16;
      // B 32x16: lane = column, 16 contiguous K values (this code's order, matching A) -> one 32B load
      v16h b = *(const v16h*)(WgT + (size_t)col * HDIM + k0 + half * 16);
      acc[t] = __builtin_amdgcn_wmma_f32_16x16x32_f16(false, a, false, b,
                                                      (short)0, acc[t], false, false);
    }
  }
  #pragma unroll
  for (int t = 0; t < 4; ++t) {
    int col = wave * 64 + t * 16 + l16;
    #pragma unroll
    for (int r = 0; r < 8; ++r)
      xh[(size_t)(row0 + r + half * 8) * XCOLS + col] = (_Float16)acc[t][r];
  }
}

// ---------------- attention scores + self-loop max init ---------------------
__global__ void __launch_bounds__(256) scores_kernel(const _Float16* __restrict__ xh,
                                                     const float* __restrict__ att_src,
                                                     const float* __restrict__ att_dst,
                                                     float* __restrict__ a_s,
                                                     float* __restrict__ a_d,
                                                     int* __restrict__ amax) {
  int gw = blockIdx.x * 8 + (threadIdx.x >> 5);   // one wave per (node, head)
  int n = gw >> 2, hh = gw & 3, lane = threadIdx.x & 31;
  if (n >= NODES) return;
  const _Float16* xp = xh + (size_t)n * XCOLS + hh * CCH + lane * 4;
  const float* as = att_src + hh * CCH + lane * 4;
  const float* ad = att_dst + hh * CCH + lane * 4;
  float ss = 0.f, sd = 0.f;
  #pragma unroll
  for (int j = 0; j < 4; ++j) { float xv = (float)xp[j]; ss += xv * as[j]; sd += xv * ad[j]; }
  ss = warp_sum(ss); sd = warp_sum(sd);
  if (lane == 0) {
    a_s[n * 4 + hh] = ss;
    a_d[n * 4 + hh] = sd;
    amax[n * 4 + hh] = fenc(lrelu(ss + sd));     // self-loop initializes the max
  }
}

// ---------------- segment max over edges (dst) ------------------------------
__global__ void edge_max_kernel(const int* __restrict__ src, const int* __restrict__ dst,
                                const float* __restrict__ a_s, const float* __restrict__ a_d,
                                int* __restrict__ amax) {
  int e = blockIdx.x * blockDim.x + threadIdx.x;
  if (e >= EDGES) return;
  int s = src[e], d = dst[e];
  #pragma unroll
  for (int hh = 0; hh < 4; ++hh) {
    float al = lrelu(a_s[s * 4 + hh] + a_d[d * 4 + hh]);
    atomicMax(&amax[d * 4 + hh], fenc(al));
  }
}

// ---------------- exp + segment denom, cache per-edge exp -------------------
__global__ void edge_expsum_kernel(const int* __restrict__ src, const int* __restrict__ dst,
                                   const float* __restrict__ a_s, const float* __restrict__ a_d,
                                   const int* __restrict__ amax,
                                   float* __restrict__ denom, float* __restrict__ wtmp) {
  int e = blockIdx.x * blockDim.x + threadIdx.x;
  if (e >= EDGES) return;
  int s = src[e], d = dst[e];
  #pragma unroll
  for (int hh = 0; hh < 4; ++hh) {
    float al = lrelu(a_s[s * 4 + hh] + a_d[d * 4 + hh]);
    float ex = expf(al - fdec(amax[d * 4 + hh]));
    wtmp[e * 4 + hh] = ex;
    atomicAdd(&denom[d * 4 + hh], ex);
  }
}

// ---------------- self-loop contribution (plain adds, own node only) --------
__global__ void __launch_bounds__(256) self_kernel(const _Float16* __restrict__ xh,
                                                   const float* __restrict__ a_s,
                                                   const float* __restrict__ a_d,
                                                   const int* __restrict__ amax,
                                                   float* __restrict__ denom,
                                                   float* __restrict__ agg) {
  int gw = blockIdx.x * 8 + (threadIdx.x >> 5);   // one wave per (node, head)
  int n = gw >> 2, hh = gw & 3, lane = threadIdx.x & 31;
  if (n >= NODES) return;
  float al = lrelu(a_s[n * 4 + hh] + a_d[n * 4 + hh]);
  float ex = expf(al - fdec(amax[n * 4 + hh]));
  if (lane == 0) denom[n * 4 + hh] += ex;
  const _Float16* xp = xh + (size_t)n * XCOLS + hh * CCH + lane * 4;
  float* ap = agg + (size_t)n * XCOLS + hh * CCH + lane * 4;
  #pragma unroll
  for (int j = 0; j < 4; ++j) ap[j] += ex * (float)xp[j];
}

// ---------------- weighted scatter: agg[dst] += ex * x[src] (L2-resident) ---
__global__ void __launch_bounds__(256) edge_agg_kernel(const int* __restrict__ src,
                                                       const int* __restrict__ dst,
                                                       const _Float16* __restrict__ xh,
                                                       const float* __restrict__ wtmp,
                                                       float* __restrict__ agg) {
  int gw = blockIdx.x * 8 + (threadIdx.x >> 5);   // one wave per (edge, head)
  int e = gw >> 2, hh = gw & 3, lane = threadIdx.x & 31;
  if (e >= EDGES) return;
  int s = src[e], d = dst[e];
  float ex = wtmp[e * 4 + hh];
  const _Float16* xp = xh + (size_t)s * XCOLS + hh * CCH + lane * 4;
  float* ap = agg + (size_t)d * XCOLS + hh * CCH + lane * 4;
  #pragma unroll
  for (int j = 0; j < 4; ++j) atomicAdd(ap + j, ex * (float)xp[j]);
}

// ---------------- fused: head-mean + FiLM + MLP(WMMA) + LN + GELU + res -----
__global__ void __launch_bounds__(256) final_kernel(const float* __restrict__ h,
                                                    const float* __restrict__ agg,
                                                    const float* __restrict__ denom,
                                                    const float* __restrict__ bias_g,
                                                    const float* __restrict__ gfilm,
                                                    const float* __restrict__ bfilm,
                                                    const _Float16* __restrict__ W1T,
                                                    const float* __restrict__ b1,
                                                    const float* __restrict__ ln_g,
                                                    const float* __restrict__ ln_b,
                                                    float* __restrict__ out) {
  __shared__ _Float16 zin[16 * 256];   // [h | sa] f16, 16 nodes
  __shared__ float    zbuf[16 * HDIM]; // MLP output tile
  const int row0 = blockIdx.x * 16, tid = threadIdx.x;

  for (int i = tid; i < 16 * HDIM; i += 256) {
    int r = i >> 7, c = i & 127, n = row0 + r;
    float am = 0.f;
    #pragma unroll
    for (int hh = 0; hh < 4; ++hh)
      am += agg[(size_t)n * XCOLS + hh * CCH + c] / denom[n * 4 + hh];
    am = 0.25f * am + bias_g[c];                 // head mean + bias
    float sa = am * (1.f + gfilm[c]) + bfilm[c]; // agg + (g*agg + b)
    zin[r * 256 + c]       = (_Float16)h[(size_t)n * HDIM + c];
    zin[r * 256 + 128 + c] = (_Float16)sa;
  }
  __syncthreads();

  const int wave = tid >> 5, lane = tid & 31, half = lane >> 4, l16 = lane & 15;
  const int col = wave * 16 + l16;               // 8 waves cover 128 output cols
  v8f acc = {};
  float bc = b1[col];
  #pragma unroll
  for (int r = 0; r < 8; ++r) acc[r] = bc;

  for (int k0 = 0; k0 < 256; k0 += 32) {
    v16h a;
    #pragma unroll
    for (int e2 = 0; e2 < 16; ++e2) {
      int k = k0 + half * 16 + e2;
      a[e2] = zin[l16 * 256 + k];
    }
    // B fragment: 16 contiguous K values per lane from transposed f16 W1
    v16h b = *(const v16h*)(W1T + (size_t)col * 256 + k0 + half * 16);
    acc = __builtin_amdgcn_wmma_f32_16x16x32_f16(false, a, false, b,
                                                 (short)0, acc, false, false);
  }
  #pragma unroll
  for (int r = 0; r < 8; ++r) zbuf[(r + half * 8) * HDIM + col] = acc[r];
  __syncthreads();

  #pragma unroll
  for (int q = 0; q < 2; ++q) {                  // each wave: 2 rows of LN
    int r = wave * 2 + q, n = row0 + r;
    float v[4];
    #pragma unroll
    for (int j = 0; j < 4; ++j) v[j] = zbuf[r * HDIM + lane * 4 + j];
    float s1 = v[0] + v[1] + v[2] + v[3];
    float s2 = v[0]*v[0] + v[1]*v[1] + v[2]*v[2] + v[3]*v[3];
    s1 = warp_sum(s1); s2 = warp_sum(s2);
    float mu   = s1 * (1.f / HDIM);
    float var  = s2 * (1.f / HDIM) - mu * mu;
    float rstd = rsqrtf(var + LN_EPS);
    #pragma unroll
    for (int j = 0; j < 4; ++j) {
      int c = lane * 4 + j;
      float zn = (v[j] - mu) * rstd * ln_g[c] + ln_b[c];
      float ge = 0.5f * zn * (1.f + erff(zn * 0.70710678118654752f));  // exact GELU
      out[(size_t)n * HDIM + c] = ge + h[(size_t)n * HDIM + c];
    }
  }
}

extern "C" void kernel_launch(void* const* d_in, const int* in_sizes, int n_in,
                              void* d_out, int out_size, void* d_ws, size_t ws_size,
                              hipStream_t stream) {
  (void)in_sizes; (void)n_in; (void)out_size; (void)ws_size;
  const float* h       = (const float*)d_in[0];
  const int*   ei      = (const int*)  d_in[1];   // [2,E]: row0=src, row1=dst
  // d_in[2] edge_weight: unused by the reference
  const float* gamma   = (const float*)d_in[3];
  const float* beta    = (const float*)d_in[4];
  const float* Wg      = (const float*)d_in[5];
  const float* att_src = (const float*)d_in[6];
  const float* att_dst = (const float*)d_in[7];
  const float* bias_g  = (const float*)d_in[8];
  const float* Wp      = (const float*)d_in[9];
  const float* bp      = (const float*)d_in[10];
  const float* W1      = (const float*)d_in[11];
  const float* b1      = (const float*)d_in[12];
  const float* ln_g    = (const float*)d_in[13];
  const float* ln_b    = (const float*)d_in[14];
  float* out = (float*)d_out;
  const int* src = ei;
  const int* dst = ei + EDGES;

  // workspace carve-up (~163.5 MB)
  char* ws = (char*)d_ws;
  size_t off = 0;
  _Float16* xh   = (_Float16*)(ws + off); off += (size_t)NODES * XCOLS * sizeof(_Float16);
  float*    a_s  = (float*)(ws + off);    off += (size_t)NODES * 4 * sizeof(float);
  float*    a_d  = (float*)(ws + off);    off += (size_t)NODES * 4 * sizeof(float);
  int*      amax = (int*)(ws + off);      off += (size_t)NODES * 4 * sizeof(int);
  float*    den  = (float*)(ws + off);    off += (size_t)NODES * 4 * sizeof(float);
  float*    agg  = (float*)(ws + off);    off += (size_t)NODES * XCOLS * sizeof(float);
  float*    wtmp = (float*)(ws + off);    off += (size_t)EDGES * 4 * sizeof(float);
  float*    gfl  = (float*)(ws + off);    off += 256 * sizeof(float);
  float*    bfl  = (float*)(ws + off);    off += 256 * sizeof(float);
  _Float16* WgT  = (_Float16*)(ws + off); off += (size_t)XCOLS * HDIM * sizeof(_Float16);
  _Float16* W1T  = (_Float16*)(ws + off); off += (size_t)HDIM * 256 * sizeof(_Float16);

  hipMemsetAsync(den, 0, (size_t)NODES * 4 * sizeof(float), stream);
  hipMemsetAsync(agg, 0, (size_t)NODES * XCOLS * sizeof(float), stream);

  film_kernel<<<1, 256, 0, stream>>>(gamma, beta, Wp, bp, gfl, bfl);
  prep_weights_kernel<<<(HDIM * XCOLS) / 256, 256, 0, stream>>>(Wg, W1, WgT, W1T);
  gemm_x_kernel<<<NODES / 16, 256, 0, stream>>>(h, WgT, xh);
  scores_kernel<<<(NODES * 4) / 8, 256, 0, stream>>>(xh, att_src, att_dst, a_s, a_d, amax);
  edge_max_kernel<<<(EDGES + 255) / 256, 256, 0, stream>>>(src, dst, a_s, a_d, amax);
  edge_expsum_kernel<<<(EDGES + 255) / 256, 256, 0, stream>>>(src, dst, a_s, a_d, amax, den, wtmp);
  self_kernel<<<(NODES * 4) / 8, 256, 0, stream>>>(xh, a_s, a_d, amax, den, agg);
  edge_agg_kernel<<<(EDGES * 4) / 8, 256, 0, stream>>>(src, dst, xh, wtmp, agg);
  final_kernel<<<NODES / 16, 256, 0, stream>>>(h, agg, den, bias_g, gfl, bfl,
                                               W1T, b1, ln_g, ln_b, out);
}
